// Attention_63548336111668
// MI455X (gfx1250) — compile-verified, matrix operand order checked by reading, not run on hardware
//
#include <hip/hip_runtime.h>
#include <hip/hip_bf16.h>

// Sliding-tile attention for MI455X (gfx1250), wave32 + WMMA bf16.
// One workgroup per (query tile, head). Flash-style online softmax.
// S^T = K * Q^T so that softmax reductions are per-lane and the P^T
// f32->bf16 pack feeds the PV WMMA B operand with no cross-lane moves.
// v2: K/V operands hoisted across query groups (3x less LDS traffic),
//     exp2-domain softmax (no log2e muls), next-tile global prefetch.

typedef __bf16 bf16_t;
typedef __attribute__((ext_vector_type(16))) __bf16 v16bf;
typedef __attribute__((ext_vector_type(8)))  __bf16 bf16x8;
typedef __attribute__((ext_vector_type(8)))  float  v8f;

#define TILE_ROWS 384   // 6*8*8
#define DHEAD     64
#define NHEADS    4

// 1/sqrt(64) * log2(e): softmax computed with exp2, scale folded into Q.
#define QSCALE 0.1803368801111204f

__device__ __forceinline__ v8f wmma_bf16(v16bf a, v16bf b, v8f c) {
  // D = A(16x32 bf16) * B(32x16 bf16) + C(16x16 f32)
  return __builtin_amdgcn_wmma_f32_16x16x32_bf16(
      /*neg_a=*/false, a, /*neg_b=*/false, b,
      /*c_mod=*/(short)0, c, /*reuse_a=*/false, /*reuse_b=*/false);
}

// Load one A/B operand (16x32 bf16, K along registers) for this lane from an
// LDS row: K = hl*8 + {0..7} and 16 + hl*8 + {0..7}  (hl = lane>>4).
__device__ __forceinline__ v16bf load_ab(const bf16_t* base, int hl) {
  union { v16bf v; bf16x8 h[2]; } u;
  u.h[0] = *(const bf16x8*)(base + hl * 8);
  u.h[1] = *(const bf16x8*)(base + 16 + hl * 8);
  return u.v;
}

__global__ __launch_bounds__(256)
void sta_attn_wmma_kernel(const float* __restrict__ q,
                          const float* __restrict__ k,
                          const float* __restrict__ v,
                          float* __restrict__ out) {
  extern __shared__ __align__(16) char smem_raw[];
  bf16_t* Qsm = (bf16_t*)smem_raw;              // [384][64], pre-scaled
  bf16_t* Ksm = Qsm + TILE_ROWS * DHEAD;        // [384][64]
  bf16_t* Vt  = Ksm + TILE_ROWS * DHEAD;        // [64][384] (transposed)

  const int qt   = blockIdx.x;    // 0..215 query tile
  const int head = blockIdx.y;    // 0..3
  const int tid  = threadIdx.x;
  const int lane = tid & 31;
  const int wv   = tid >> 5;      // wave id 0..7
  const int hl   = lane >> 4;     // lane half
  const int lrow = lane & 15;

  const int nt = qt / 36, nh = (qt / 6) % 6, nw = qt % 6;

  // Key-tile window per head (WINDOWS = (2,1,1),(1,2,2),(1,1,2),(1,1,1))
  int W, kt[4];
  if (head == 0) {
    int ts = nt < 4 ? nt : 4;  W = 2;
    kt[0] = (ts * 6 + nh) * 6 + nw;  kt[1] = ((ts + 1) * 6 + nh) * 6 + nw;
    kt[2] = kt[3] = kt[0];
  } else if (head == 1) {
    int hs = nh < 4 ? nh : 4, ws = nw < 4 ? nw : 4;  W = 4;
    kt[0] = (nt * 6 + hs) * 6 + ws;       kt[1] = kt[0] + 1;
    kt[2] = (nt * 6 + hs + 1) * 6 + ws;   kt[3] = kt[2] + 1;
  } else if (head == 2) {
    int ws = nw < 4 ? nw : 4;  W = 2;
    kt[0] = (nt * 6 + nh) * 6 + ws;  kt[1] = kt[0] + 1;
    kt[2] = kt[3] = kt[0];
  } else {
    W = 1;  kt[0] = kt[1] = kt[2] = kt[3] = qt;
  }

  // ---- load Q tile into LDS (fp32 -> bf16, scale folded, exp2 domain) ----
  {
    const int sub = tid & 3, r0 = tid >> 2;
    for (int it = 0; it < 6; ++it) {
      int r = r0 + it * 64;
      int tt = r >> 6, th = (r >> 3) & 7, tw = r & 7;
      long s = ((((long)(nt * 6 + tt) * 6 + nh) * 8 + th) * 6 + nw) * 8 + tw;
      const float4* g = (const float4*)(q + (s * NHEADS + head) * DHEAD + sub * 16);
      bf16_t* dst = Qsm + r * DHEAD + sub * 16;
#pragma unroll
      for (int jj = 0; jj < 4; ++jj) {
        float4 x = g[jj];
        dst[jj * 4 + 0] = (bf16_t)(x.x * QSCALE);
        dst[jj * 4 + 1] = (bf16_t)(x.y * QSCALE);
        dst[jj * 4 + 2] = (bf16_t)(x.z * QSCALE);
        dst[jj * 4 + 3] = (bf16_t)(x.w * QSCALE);
      }
    }
  }
  __syncthreads();

  // Q^T B-operands, held in registers for the whole kernel.
  v16bf qB[3][2];
#pragma unroll
  for (int g = 0; g < 3; ++g) {
    const bf16_t* row = Qsm + ((wv * 3 + g) * 16 + lrow) * DHEAD;
    qB[g][0] = load_ab(row, hl);
    qB[g][1] = load_ab(row + 32, hl);
  }

  // O^T accumulators (4 d-tiles x 16 queries each), running max / sum.
  v8f  O[3][4];
  float m_[3], l_[3];
  const v8f vzero = {};
#pragma unroll
  for (int g = 0; g < 3; ++g) {
    m_[g] = -3.0e38f; l_[g] = 0.0f;
#pragma unroll
    for (int t = 0; t < 4; ++t) O[g][t] = vzero;
  }

  for (int j = 0; j < W; ++j) {
    __syncthreads();  // prior compute done before LDS overwrite
    // ---- cooperative load of K tile (row-major) and V tile (transposed) ----
    {
      const int T = kt[j];
      const int ktt = T / 36, kkh = (T / 6) % 6, kkw = T % 6;
      const int sub = tid & 3, r0 = tid >> 2;
      for (int it = 0; it < 6; ++it) {
        int r = r0 + it * 64;
        int tt = r >> 6, th = (r >> 3) & 7, tw = r & 7;
        long s = ((((long)(ktt * 6 + tt) * 6 + kkh) * 8 + th) * 6 + kkw) * 8 + tw;
        long base = (s * NHEADS + head) * DHEAD + sub * 16;
        const float4* gk = (const float4*)(k + base);
        const float4* gv = (const float4*)(v + base);
        bf16_t* dk = Ksm + r * DHEAD + sub * 16;
#pragma unroll
        for (int jj = 0; jj < 4; ++jj) {
          float4 x = gk[jj];
          dk[jj * 4 + 0] = (bf16_t)x.x; dk[jj * 4 + 1] = (bf16_t)x.y;
          dk[jj * 4 + 2] = (bf16_t)x.z; dk[jj * 4 + 3] = (bf16_t)x.w;
          float4 y = gv[jj];
          int c0 = sub * 16 + jj * 4;
          Vt[(c0 + 0) * TILE_ROWS + r] = (bf16_t)y.x;
          Vt[(c0 + 1) * TILE_ROWS + r] = (bf16_t)y.y;
          Vt[(c0 + 2) * TILE_ROWS + r] = (bf16_t)y.z;
          Vt[(c0 + 3) * TILE_ROWS + r] = (bf16_t)y.w;
        }
      }
    }
    __syncthreads();

    // ---- prefetch next key tile into L2 while we compute this one ----
    if (j + 1 < W) {
      const int T = kt[j + 1];
      const int ktt = T / 36, kkh = (T / 6) % 6, kkw = T % 6;
      const int sub = tid & 3, r0 = tid >> 2;
      for (int it = 0; it < 6; ++it) {
        int r = r0 + it * 64;
        int tt = r >> 6, th = (r >> 3) & 7, tw = r & 7;
        long s = ((((long)(ktt * 6 + tt) * 6 + kkh) * 8 + th) * 6 + kkw) * 8 + tw;
        long base = (s * NHEADS + head) * DHEAD + sub * 16;
        __builtin_prefetch(k + base, 0, 1);
        __builtin_prefetch(v + base, 0, 1);
      }
    }

    // ---- flash-attention over this key tile, 32-key chunks ----
    for (int c = 0; c < 12; ++c) {
      const int kb = c * 32;
      // K/V operands loaded ONCE per chunk, reused by all 3 query groups.
      const bf16_t* kr0 = Ksm + (kb + lrow) * DHEAD;
      const bf16_t* kr1 = kr0 + 16 * DHEAD;
      v16bf kA00 = load_ab(kr0,      hl);
      v16bf kA01 = load_ab(kr0 + 32, hl);
      v16bf kA10 = load_ab(kr1,      hl);
      v16bf kA11 = load_ab(kr1 + 32, hl);
      v16bf vA[4];
#pragma unroll
      for (int t = 0; t < 4; ++t)
        vA[t] = load_ab(Vt + (t * 16 + lrow) * TILE_ROWS + kb, hl);

#pragma unroll
      for (int g = 0; g < 3; ++g) {
        v8f s0 = vzero, s1 = vzero;
        s0 = wmma_bf16(kA00, qB[g][0], s0);
        s0 = wmma_bf16(kA01, qB[g][1], s0);
        s1 = wmma_bf16(kA10, qB[g][0], s1);
        s1 = wmma_bf16(kA11, qB[g][1], s1);

        // per-query chunk max (lane = query; combine row halves via xor-16)
        float mc = s0[0];
#pragma unroll
        for (int e = 1; e < 8; ++e) mc = fmaxf(mc, s0[e]);
#pragma unroll
        for (int e = 0; e < 8; ++e) mc = fmaxf(mc, s1[e]);
        mc = fmaxf(mc, __shfl_xor(mc, 16, 32));

        float mn    = fmaxf(m_[g], mc);
        float alpha = __builtin_amdgcn_exp2f(m_[g] - mn);

        float rs = 0.0f;
        union { v16bf v; bf16x8 h[2]; } pb;  // P^T packed as PV B-operand
#pragma unroll
        for (int e = 0; e < 8; ++e) {
          float p0 = __builtin_amdgcn_exp2f(s0[e] - mn);
          float p1 = __builtin_amdgcn_exp2f(s1[e] - mn);
          rs += p0 + p1;
          pb.v[e]     = (bf16_t)p0;
          pb.v[e + 8] = (bf16_t)p1;
        }
        rs += __shfl_xor(rs, 16, 32);
        l_[g] = l_[g] * alpha + rs;
        m_[g] = mn;

#pragma unroll
        for (int t = 0; t < 4; ++t) {
#pragma unroll
          for (int e = 0; e < 8; ++e) O[g][t][e] *= alpha;
          O[g][t] = wmma_bf16(vA[t], pb.v, O[g][t]);
        }
      }
    }
  }

  // ---- normalize and store (untile fold: natural-layout addressing) ----
#pragma unroll
  for (int g = 0; g < 3; ++g) {
    float inv = 1.0f / l_[g];
    int qrow = (wv * 3 + g) * 16 + lrow;
    int tt = qrow >> 6, th = (qrow >> 3) & 7, tw = qrow & 7;
    long s = ((((long)(nt * 6 + tt) * 6 + nh) * 8 + th) * 6 + nw) * 8 + tw;
    float* op = out + (s * NHEADS + head) * DHEAD + hl * 8;
#pragma unroll
    for (int t = 0; t < 4; ++t)
#pragma unroll
      for (int e = 0; e < 8; ++e)
        op[t * 16 + e] = O[g][t][e] * inv;
  }
}

extern "C" void kernel_launch(void* const* d_in, const int* in_sizes, int n_in,
                              void* d_out, int out_size, void* d_ws, size_t ws_size,
                              hipStream_t stream) {
  const float* q = (const float*)d_in[0];
  const float* k = (const float*)d_in[1];
  const float* v = (const float*)d_in[2];
  float* out = (float*)d_out;
  (void)in_sizes; (void)n_in; (void)out_size; (void)d_ws; (void)ws_size;

  dim3 grid(216, 4);                                   // (query tile, head)
  size_t lds = (size_t)3 * TILE_ROWS * DHEAD * sizeof(bf16_t);  // 144 KB
  sta_attn_wmma_kernel<<<grid, dim3(256), lds, stream>>>(q, k, v, out);
}
